// QAConv_33225867002568
// MI455X (gfx1250) — matrix operand order repaired and checked by reading, not compiled, not run on hardware
//
#include <hip/hip_runtime.h>
#include <hip/hip_bf16.h>
#include <math.h>

typedef __attribute__((ext_vector_type(16))) _Float16 v16h;
typedef __attribute__((ext_vector_type(8)))  _Float16 v8h;
typedef __attribute__((ext_vector_type(8)))  float    v8f;
typedef __attribute__((ext_vector_type(4)))  unsigned int v4u;
typedef __attribute__((ext_vector_type(4)))  int v4i;
typedef __attribute__((ext_vector_type(8)))  int v8i;

#define NG 64
#define NP 64
#define ND 512
#define NHW 192
#define KC 64            // K-chunk halfs staged per TDM transfer
#define LDS_STRIDE 72    // 64 + 8-half TDM pad -> conflict-free ds_load_b128
#define EPS_BN 1e-5f

// ---------------------------------------------------------------------------
// Order-preserving float <-> uint map so float-max == uint-max (ds_max_u32).
// ---------------------------------------------------------------------------
__device__ __forceinline__ unsigned f2ord(float f) {
    unsigned u = __float_as_uint(f);
    return (u & 0x80000000u) ? ~u : (u | 0x80000000u);
}
__device__ __forceinline__ float ord2f(unsigned u) {
    return __uint_as_float((u & 0x80000000u) ? (u & 0x7fffffffu) : ~u);
}

// ---------------------------------------------------------------------------
// Pack kernel: L2-normalize over channel dim and transpose to [n][pos][d] f16.
// One wave32 per (tensor-index, spatial-position).
// ---------------------------------------------------------------------------
__global__ __launch_bounds__(256) void qaconv_pack(const float* __restrict__ in,
                                                   _Float16* __restrict__ out)
{
    const int wid  = blockIdx.x * 8 + (threadIdx.x >> 5);
    const int lane = threadIdx.x & 31;
    const int t    = wid / NHW;
    const int s    = wid - t * NHW;

    const float* src = in + (size_t)t * ND * NHW + s;
    float vals[16];
    float ss = 0.f;
#pragma unroll
    for (int k = 0; k < 16; ++k) {
        float v = src[(size_t)(lane + k * 32) * NHW];
        vals[k] = v;
        ss += v * v;
    }
#pragma unroll
    for (int m = 16; m >= 1; m >>= 1) ss += __shfl_xor(ss, m);
    const float rn = 1.0f / fmaxf(sqrtf(ss), 1e-12f);

    _Float16* dst = out + (size_t)t * NHW * ND + (size_t)s * ND;
#pragma unroll
    for (int k = 0; k < 16; ++k)
        dst[lane + k * 32] = (_Float16)(vals[k] * rn);
}

// ---------------------------------------------------------------------------
// Tensor Data Mover: stage a [192 rows x 64 halfs] tile (row stride 512 halfs)
// from global into LDS, with 4-DWORD padding every 32 DWORDs so the LDS row
// stride becomes 72 halfs (144 B) -> bank-conflict-free fragment loads.
// ---------------------------------------------------------------------------
__device__ __forceinline__ unsigned ldsOffset(const void* p) {
    // flat LDS aperture: addr[31:0] is the LDS byte offset
    return (unsigned)(unsigned long long)p;
}

__device__ __forceinline__ void tdm_load_tile(const _Float16* gsrc, unsigned lds_off)
{
    const unsigned long long ga = (unsigned long long)(const void*)gsrc;
    v4u g0;
    g0.x = 1u;                                    // count=1 (valid user D#)
    g0.y = lds_off;                               // lds_addr [63:32]
    g0.z = (unsigned)ga;                          // global_addr lo
    g0.w = (unsigned)(ga >> 32) | 0x80000000u;    // global_addr hi | type=2
    v8i g1;
    g1[0] = (int)((1u << 16)        // data_size = 2 bytes
                | (1u << 20)        // pad_enable
                | (4u << 22)        // pad_interval: 32 DWORDs
                | (3u << 25));      // pad_amount : 4 DWORDs
    g1[1] = (int)((unsigned)ND << 16);    // tensor_dim0 = 512 (bits 79:48)
    g1[2] = (int)((unsigned)NHW << 16);   // tensor_dim1 = 192 (bits 111:80)
    g1[3] = (int)((unsigned)KC << 16);    // tile_dim0   = 64  (bits 127:112)
    g1[4] = NHW;                          // tile_dim1 = 192, tile_dim2 = 0
    g1[5] = ND;                           // tensor_dim0_stride = 512
    g1[6] = 0;
    g1[7] = 0;
    v4i z4 = {0, 0, 0, 0};
#if __has_include(<hip/amd_detail/amd_gfx1250_TDM.h>)
    v8i z8 = {0, 0, 0, 0, 0, 0, 0, 0};
    __builtin_amdgcn_tensor_load_to_lds(g0, g1, z4, z4, z8, 0);   // 6-arg toolchain
#else
    __builtin_amdgcn_tensor_load_to_lds(g0, g1, z4, z4, 0);       // 5-arg toolchain
#endif
}

// ---------------------------------------------------------------------------
// LDS fragment loaders matching CDNA5 16-bit WMMA VGPR layouts (wave32).
// ---------------------------------------------------------------------------
union Frag { v16h v; v8h h[2]; };

__device__ __forceinline__ v16h lds_a(const _Float16* base, int row, int ks, int hi) {
    const _Float16* p = base + row * LDS_STRIDE + ks + hi * 8;
    Frag f;
    f.h[0] = *(const v8h*)p;          // K ks+hi*8 .. +7
    f.h[1] = *(const v8h*)(p + 16);   // K ks+hi*8+16 .. +23
    return f.v;
}
__device__ __forceinline__ v16h lds_b(const _Float16* base, int row, int ks, int hi) {
    const _Float16* p = base + row * LDS_STRIDE + ks + hi * 8;
    Frag f;
    f.h[0] = *(const v8h*)p;          // K ks+hi*8 .. +7
    f.h[1] = *(const v8h*)(p + 16);   // K ks+hi*8+16 .. +23
    return f.v;
}

// Fold one 16x16 f32 accumulator tile into the running max arrays.
// C/D layout: lane l, vgpr v -> element (M = 8*(l>>4)+v, N = l&15).
__device__ __forceinline__ void reduce_tile(const v8f c,
                                            unsigned* __restrict__ maxI,
                                            unsigned* __restrict__ maxJ,
                                            int ibase, int jbase, int nn, int hi)
{
    float r = c[0];
#pragma unroll
    for (int v = 1; v < 8; ++v) r = fmaxf(r, c[v]);
    atomicMax(&maxI[jbase + nn], f2ord(r));       // max over i, per column j

#pragma unroll
    for (int v = 0; v < 8; ++v) {
        float t = c[v];
        t = fmaxf(t, __shfl_xor(t, 1));
        t = fmaxf(t, __shfl_xor(t, 2));
        t = fmaxf(t, __shfl_xor(t, 4));
        t = fmaxf(t, __shfl_xor(t, 8));           // xor<16 stays in 16-lane half
        if (nn == 0)
            atomicMax(&maxJ[ibase + hi * 8 + v], f2ord(t));  // max over j, per row i
    }
}

// ---------------------------------------------------------------------------
// Main kernel: one block (8 waves) per (g,p). Double-buffered TDM staging of
// K-chunks into LDS; each wave owns a 48x96 output region (3x6 WMMA tiles,
// persistent accumulators); fused dual max-pool + BN + fc + BN + sigmoid.
// ---------------------------------------------------------------------------
__global__ __launch_bounds__(256) void qaconv_main(
    const _Float16* __restrict__ ker,   // [P][HW][D]  (probe, i-major)
    const _Float16* __restrict__ gal,   // [G][HW][D]  (gallery, j-major)
    const float* __restrict__ fcw,      // [2*HW]
    const float* __restrict__ bn_w,  const float* __restrict__ bn_b,
    const float* __restrict__ bn_m,  const float* __restrict__ bn_v,
    const float* __restrict__ fc_b,
    const float* __restrict__ lbn_w, const float* __restrict__ lbn_b,
    const float* __restrict__ lbn_m, const float* __restrict__ lbn_v,
    float* __restrict__ out)
{
    __shared__ __attribute__((aligned(16))) _Float16 As[2][NHW * LDS_STRIDE];
    __shared__ __attribute__((aligned(16))) _Float16 Bs[2][NHW * LDS_STRIDE];
    __shared__ unsigned maxI[NHW];   // max over i, indexed by j
    __shared__ unsigned maxJ[NHW];   // max over j, indexed by i
    __shared__ float red[256];
    __shared__ float redw[256];

    const int gp   = blockIdx.x;
    const int gi   = gp >> 6;
    const int pi   = gp & 63;
    const int tid  = threadIdx.x;
    const int lane = tid & 31;
    const int wave = tid >> 5;
    const int nn   = lane & 15;
    const int hi   = lane >> 4;

    if (tid < NHW) {
        const unsigned NEGINF = f2ord(-3.0e38f);
        maxI[tid] = NEGINF;
        maxJ[tid] = NEGINF;
    }

    const _Float16* kerp = ker + (size_t)pi * NHW * ND;
    const _Float16* galg = gal + (size_t)gi * NHW * ND;

    // wave's 48x96 region of the 192x192 similarity matrix (4x2 grid of waves)
    const int i0 = (wave >> 1) * 48;
    const int j0 = (wave & 1) * 96;

    v8f acc[3][6];
#pragma unroll
    for (int r = 0; r < 3; ++r)
#pragma unroll
        for (int q = 0; q < 6; ++q) acc[r][q] = (v8f){};

    // prologue: stage chunk 0 into buffer 0 (wave 0: A, wave 1: B)
    if (wave == 0)      tdm_load_tile(kerp, ldsOffset(&As[0][0]));
    else if (wave == 1) tdm_load_tile(galg, ldsOffset(&Bs[0][0]));

    for (int c = 0; c < ND / KC; ++c) {
        const int buf = c & 1;
        if (c + 1 < ND / KC) {
            if (wave == 0)      tdm_load_tile(kerp + (c + 1) * KC, ldsOffset(&As[buf ^ 1][0]));
            else if (wave == 1) tdm_load_tile(galg + (c + 1) * KC, ldsOffset(&Bs[buf ^ 1][0]));
            __builtin_amdgcn_s_wait_tensorcnt(1);   // oldest (chunk c) complete
        } else {
            __builtin_amdgcn_s_wait_tensorcnt(0);
        }
        __syncthreads();                            // chunk c visible to all waves

        const _Float16* Ab = &As[buf][0];
        const _Float16* Bb = &Bs[buf][0];
#pragma unroll
        for (int ks = 0; ks < KC; ks += 32) {
            v16h a[3], b[6];
#pragma unroll
            for (int r = 0; r < 3; ++r) a[r] = lds_a(Ab, i0 + 16 * r + nn, ks, hi);
#pragma unroll
            for (int q = 0; q < 6; ++q) b[q] = lds_b(Bb, j0 + 16 * q + nn, ks, hi);
#pragma unroll
            for (int r = 0; r < 3; ++r)
#pragma unroll
                for (int q = 0; q < 6; ++q)
                    acc[r][q] = __builtin_amdgcn_wmma_f32_16x16x32_f16(
                        false, a[r], false, b[q], (short)0, acc[r][q], false, false);
        }
        __syncthreads();   // everyone done reading buf before it is overwritten
    }

    // fold this wave's 18 tiles into the shared max arrays
#pragma unroll
    for (int r = 0; r < 3; ++r)
#pragma unroll
        for (int q = 0; q < 6; ++q)
            reduce_tile(acc[r][q], maxI, maxJ, i0 + 16 * r, j0 + 16 * q, nn, hi);
    __syncthreads();

    // fc dot product with scalar BN folded in: fc = a*dot + b*sumW + fc_bias
    float pd = 0.f, pw = 0.f;
    if (tid < NHW) {
        const float mi = ord2f(maxI[tid]);
        const float mj = ord2f(maxJ[tid]);
        const float w0 = fcw[tid];
        const float w1 = fcw[NHW + tid];
        pd = mi * w0 + mj * w1;
        pw = w0 + w1;
    }
    red[tid]  = pd;
    redw[tid] = pw;
    __syncthreads();
#pragma unroll
    for (int s = 128; s >= 1; s >>= 1) {
        if (tid < s) { red[tid] += red[tid + s]; redw[tid] += redw[tid + s]; }
        __syncthreads();
    }

    if (tid == 0) {
        const float a   = bn_w[0] * rsqrtf(bn_v[0] + EPS_BN);
        const float b   = bn_b[0] - bn_m[0] * a;
        const float fco = a * red[0] + b * redw[0] + fc_b[0];
        const float la  = lbn_w[0] * rsqrtf(lbn_v[0] + EPS_BN);
        const float y   = (fco - lbn_m[0]) * la + lbn_b[0];
        out[gp] = 1.0f / (1.0f + __expf(-y * 0.1f));
    }
}

// ---------------------------------------------------------------------------
extern "C" void kernel_launch(void* const* d_in, const int* in_sizes, int n_in,
                              void* d_out, int out_size, void* d_ws, size_t ws_size,
                              hipStream_t stream)
{
    const float* gal_fea  = (const float*)d_in[0];
    const float* prob_fea = (const float*)d_in[1];
    const float* bn_w     = (const float*)d_in[2];
    const float* bn_b     = (const float*)d_in[3];
    const float* bn_m     = (const float*)d_in[4];
    const float* bn_v     = (const float*)d_in[5];
    const float* fc_w     = (const float*)d_in[6];
    const float* fc_b     = (const float*)d_in[7];
    const float* lbn_w    = (const float*)d_in[8];
    const float* lbn_b    = (const float*)d_in[9];
    const float* lbn_m    = (const float*)d_in[10];
    const float* lbn_v    = (const float*)d_in[11];
    float* out            = (float*)d_out;

    const size_t packBytes = (size_t)NP * NHW * ND * sizeof(_Float16);
    _Float16* ker_ws = (_Float16*)d_ws;                       // [P][HW][D]
    _Float16* gal_ws = (_Float16*)((char*)d_ws + packBytes);  // [G][HW][D]

    qaconv_pack<<<(NP * NHW) / 8, 256, 0, stream>>>(prob_fea, ker_ws);
    qaconv_pack<<<(NG * NHW) / 8, 256, 0, stream>>>(gal_fea,  gal_ws);

    qaconv_main<<<NG * NP, 256, 0, stream>>>(ker_ws, gal_ws, fc_w,
                                             bn_w, bn_b, bn_m, bn_v, fc_b,
                                             lbn_w, lbn_b, lbn_m, lbn_v, out);
}
